// NodeMLP_49263274885775
// MI455X (gfx1250) — compile-verified, matrix operand order checked by reading, not run on hardware
//
#include <hip/hip_runtime.h>

#define N_PTS 1000000
#define G_GR  16
#define D0    32
#define D1    128
#define D2    64
#define M_SUP 50000
#define EPSV  1e-5f
#define SLOPE 0.01f
#define LDA   136   // padded bf16 LDS row stride (272B) to avoid bank conflicts

typedef __bf16 bf16;
typedef bf16  v16bf __attribute__((ext_vector_type(16)));
typedef bf16  v8bf  __attribute__((ext_vector_type(8)));
typedef float v8f   __attribute__((ext_vector_type(8)));

__device__ __forceinline__ v8f wmma_bf16(v16bf a, v16bf b, v8f c) {
  return __builtin_amdgcn_wmma_f32_16x16x32_bf16(false, a, false, b, (short)0, c,
                                                 false, false);
}

__device__ __forceinline__ v16bf zero_frag() {
  v16bf z;
#pragma unroll
  for (int i = 0; i < 16; ++i) z[i] = (bf16)0.0f;
  return z;
}

// 16-bit A/B fragment from 24 consecutive floats:
// elements 0..7 <- s[0..7], elements 8..15 <- s[16..23]  (ISA 7.12.2 layout,
// caller passes s = row_base + (lane&16 ? 8 : 0))
__device__ __forceinline__ v16bf load_frag_f32(const float* __restrict__ s) {
  const float4* p = (const float4*)s;
  float4 a0 = p[0], a1 = p[1], b0 = p[4], b1 = p[5];
  v16bf r;
  r[0]=(bf16)a0.x;  r[1]=(bf16)a0.y;  r[2]=(bf16)a0.z;  r[3]=(bf16)a0.w;
  r[4]=(bf16)a1.x;  r[5]=(bf16)a1.y;  r[6]=(bf16)a1.z;  r[7]=(bf16)a1.w;
  r[8]=(bf16)b0.x;  r[9]=(bf16)b0.y;  r[10]=(bf16)b0.z; r[11]=(bf16)b0.w;
  r[12]=(bf16)b1.x; r[13]=(bf16)b1.y; r[14]=(bf16)b1.z; r[15]=(bf16)b1.w;
  return r;
}

__device__ __forceinline__ v16bf load_frag_lds(const bf16* s) {
  v8bf lo = *(const v8bf*)s;
  v8bf hi = *(const v8bf*)(s + 16);
  v16bf r;
#pragma unroll
  for (int i = 0; i < 8; ++i) { r[i] = lo[i]; r[i + 8] = hi[i]; }
  return r;
}

__device__ __forceinline__ float leaky(float v) {
  return (v > 0.0f) ? v : SLOPE * v;
}

// order-preserving float -> u32 key (so segment-max uses native u32 atomic max)
__device__ __forceinline__ unsigned fkey(float f) {
  unsigned u = __float_as_uint(f);
  return (u & 0x80000000u) ? ~u : (u | 0x80000000u);
}
__device__ __forceinline__ float funkey(unsigned k) {
  return (k & 0x80000000u) ? __uint_as_float(k ^ 0x80000000u)
                           : __uint_as_float(~k);
}

__device__ __forceinline__ int lower_bound_i32(const int* __restrict__ a, int n, int v) {
  int lo = 0, hi = n;
  while (lo < hi) { int mid = (lo + hi) >> 1; if (a[mid] < v) lo = mid + 1; else hi = mid; }
  return lo;
}

// ---------------------------------------------------------------------------
// init: zero stat accumulators, set output keys to key(-inf)
// ---------------------------------------------------------------------------
__global__ __launch_bounds__(256) void k_init(unsigned* __restrict__ outk,
                                              float* __restrict__ ws) {
  int i = blockIdx.x * 256 + threadIdx.x;
  if (i < M_SUP * D2) outk[i] = 0x007FFFFFu;           // fkey(-inf)
  if (i < 2 * G_GR * D1 + 2 * G_GR * D2) ws[i] = 0.0f; // sum1,sq1,sum2,sq2
}

// ---------------------------------------------------------------------------
// pass 1: h1 = x@W1^T + b1 via bf16 WMMA; per-graph sum / sumsq of h1
// block = 256 threads = 8 waves; tile = 128 nodes x 128 channels, K = 32
// ---------------------------------------------------------------------------
__global__ __launch_bounds__(256) void k_stats1(const float* __restrict__ x,
                                                const float* __restrict__ W1,
                                                const float* __restrict__ b1,
                                                const int* __restrict__ gidx,
                                                float* __restrict__ sum1,
                                                float* __restrict__ sq1) {
  __shared__ float s_sum[D1];
  __shared__ float s_sq[D1];
  __shared__ int   s_g[128];
  const int tid  = threadIdx.x;
  const int wave = tid >> 5;
  const int lane = tid & 31;
  const int row0 = blockIdx.x * 128;

  if (tid < D1) {
    s_sum[tid] = 0.0f;
    s_sq[tid]  = 0.0f;
    int row = row0 + tid;
    s_g[tid] = (row < N_PTS) ? gidx[row] : 0;
  }
  __syncthreads();

  const int  gfirst = s_g[0];
  const bool fast   = (s_g[min(127, N_PTS - 1 - row0)] == gfirst);

  const int kb   = (lane & 16) ? 8 : 0;
  const int arow = row0 + wave * 16 + (lane & 15);
  v16bf afrag = (arow < N_PTS) ? load_frag_f32(x + arow * D0 + kb) : zero_frag();

  const int lrb = wave * 16 + ((lane & 16) ? 8 : 0);   // local row base of C frag

  for (int tn = 0; tn < 8; ++tn) {
    const int bcol = tn * 16 + (lane & 15);
    v16bf bfrag = load_frag_f32(W1 + bcol * D0 + kb);
    v8f c = {0.f, 0.f, 0.f, 0.f, 0.f, 0.f, 0.f, 0.f};
    c = wmma_bf16(afrag, bfrag, c);
    const float bias = b1[bcol];
    if (fast) {
      float cs = 0.0f, cq = 0.0f;
#pragma unroll
      for (int r = 0; r < 8; ++r) {
        if (row0 + lrb + r < N_PTS) {
          float v = c[r] + bias;
          cs += v; cq += v * v;
        }
      }
      atomicAdd(&s_sum[bcol], cs);
      atomicAdd(&s_sq[bcol], cq);
    } else {
#pragma unroll
      for (int r = 0; r < 8; ++r) {
        if (row0 + lrb + r < N_PTS) {
          float v = c[r] + bias;
          int g = s_g[lrb + r];
          unsafeAtomicAdd(&sum1[g * D1 + bcol], v);
          unsafeAtomicAdd(&sq1[g * D1 + bcol], v * v);
        }
      }
    }
  }
  __syncthreads();
  if (fast && tid < D1) {
    unsafeAtomicAdd(&sum1[gfirst * D1 + tid], s_sum[tid]);
    unsafeAtomicAdd(&sq1[gfirst * D1 + tid], s_sq[tid]);
  }
}

// ---------------------------------------------------------------------------
// finalize stats -> per (graph,channel) fused scale/shift:
//   a = leaky( h*A + B ),  A = rstd*gamma,  B = beta - mean*rstd*gamma
// counts via binary search on the *sorted* norm_index
// ---------------------------------------------------------------------------
__global__ __launch_bounds__(256) void k_finalize(const float* __restrict__ sum,
                                                  const float* __restrict__ sq,
                                                  const float* __restrict__ gamma,
                                                  const float* __restrict__ beta,
                                                  const int* __restrict__ gidx,
                                                  float* __restrict__ Ac,
                                                  float* __restrict__ Bc,
                                                  int d) {
  int t = blockIdx.x * 256 + threadIdx.x;
  if (t >= G_GR * d) return;
  int g = t / d, ch = t % d;
  int lo = lower_bound_i32(gidx, N_PTS, g);
  int hi = lower_bound_i32(gidx, N_PTS, g + 1);
  float cnt  = (float)max(hi - lo, 1);
  float mean = sum[t] / cnt;
  float var  = fmaxf(sq[t] / cnt - mean * mean, 0.0f);
  float rstd = rsqrtf(var + EPSV);
  float sc   = rstd * gamma[ch];
  Ac[t] = sc;
  Bc[t] = beta[ch] - mean * sc;
}

// ---------------------------------------------------------------------------
// fused pass: recompute h1, norm1+leaky -> LDS bf16 tile, layer-2 WMMA (K=128)
//   FINAL=false: accumulate per-graph sum/sumsq of h2
//   FINAL=true : norm2+leaky and scatter-max into superpoints (u32-key atomics)
// ---------------------------------------------------------------------------
template <bool FINAL>
__global__ __launch_bounds__(256) void k_fused(const float* __restrict__ x,
                                               const float* __restrict__ W1,
                                               const float* __restrict__ b1,
                                               const float* __restrict__ W2,
                                               const float* __restrict__ b2,
                                               const int* __restrict__ gidx,
                                               const int* __restrict__ sidx,
                                               const float* __restrict__ A1c,
                                               const float* __restrict__ B1c,
                                               const float* __restrict__ A2c,
                                               const float* __restrict__ B2c,
                                               float* __restrict__ sum2,
                                               float* __restrict__ sq2,
                                               unsigned* __restrict__ outk) {
  __shared__ bf16  a1s[128 * LDA];
  __shared__ float s_sum[D2];
  __shared__ float s_sq[D2];
  __shared__ int   s_g[128];
  __shared__ int   s_s[128];

  const int tid  = threadIdx.x;
  const int wave = tid >> 5;
  const int lane = tid & 31;
  const int row0 = blockIdx.x * 128;

  if (tid < 128) {
    if (!FINAL && tid < D2) { s_sum[tid] = 0.0f; s_sq[tid] = 0.0f; }
    int row = row0 + tid;
    s_g[tid] = (row < N_PTS) ? gidx[row] : 0;
    if (FINAL) s_s[tid] = (row < N_PTS) ? sidx[row] : 0;
  }
  __syncthreads();

  const int  gfirst = s_g[0];
  const bool fast   = (s_g[min(127, N_PTS - 1 - row0)] == gfirst);

  const int kb   = (lane & 16) ? 8 : 0;
  const int arow = row0 + wave * 16 + (lane & 15);
  const int lrb  = wave * 16 + ((lane & 16) ? 8 : 0);   // local row base of C frag
  v16bf afrag = (arow < N_PTS) ? load_frag_f32(x + arow * D0 + kb) : zero_frag();

  // ---- layer 1: h1 tile -> normalized bf16 activations in LDS
  for (int tn = 0; tn < 8; ++tn) {
    const int bcol = tn * 16 + (lane & 15);
    v16bf bfrag = load_frag_f32(W1 + bcol * D0 + kb);
    v8f c = {0.f, 0.f, 0.f, 0.f, 0.f, 0.f, 0.f, 0.f};
    c = wmma_bf16(afrag, bfrag, c);
    const float bias = b1[bcol];
    if (fast) {
      const float sc = A1c[gfirst * D1 + bcol];
      const float sh = B1c[gfirst * D1 + bcol];
#pragma unroll
      for (int r = 0; r < 8; ++r) {
        int lr = lrb + r;
        float v = (row0 + lr < N_PTS) ? leaky(fmaf(c[r] + bias, sc, sh)) : 0.0f;
        a1s[lr * LDA + bcol] = (bf16)v;
      }
    } else {
#pragma unroll
      for (int r = 0; r < 8; ++r) {
        int lr = lrb + r;
        float v = 0.0f;
        if (row0 + lr < N_PTS) {
          int g = s_g[lr];
          v = leaky(fmaf(c[r] + bias, A1c[g * D1 + bcol], B1c[g * D1 + bcol]));
        }
        a1s[lr * LDA + bcol] = (bf16)v;
      }
    }
  }
  __syncthreads();

  // ---- layer 2: K = 128 -> 4 chained WMMAs per 16x16 output tile
  const int lr2 = wave * 16 + (lane & 15);
  for (int tn2 = 0; tn2 < 4; ++tn2) {
    const int col2 = tn2 * 16 + (lane & 15);
    v8f c2 = {0.f, 0.f, 0.f, 0.f, 0.f, 0.f, 0.f, 0.f};
#pragma unroll
    for (int kk = 0; kk < 4; ++kk) {
      v16bf a2  = load_frag_lds(&a1s[lr2 * LDA + kk * 32 + kb]);
      v16bf b2f = load_frag_f32(W2 + col2 * D1 + kk * 32 + kb);
      c2 = wmma_bf16(a2, b2f, c2);
    }
    const float bias2 = b2[col2];
    if (FINAL) {
      const float scf = A2c[gfirst * D2 + col2];
      const float shf = B2c[gfirst * D2 + col2];
#pragma unroll
      for (int r = 0; r < 8; ++r) {
        int lr = lrb + r;
        if (row0 + lr < N_PTS) {
          float sc = scf, sh = shf;
          if (!fast) {
            int g = s_g[lr];
            sc = A2c[g * D2 + col2];
            sh = B2c[g * D2 + col2];
          }
          float v = leaky(fmaf(c2[r] + bias2, sc, sh));
          atomicMax(&outk[s_s[lr] * D2 + col2], fkey(v));
        }
      }
    } else if (fast) {
      float cs = 0.0f, cq = 0.0f;
#pragma unroll
      for (int r = 0; r < 8; ++r) {
        if (row0 + lrb + r < N_PTS) {
          float v = c2[r] + bias2;
          cs += v; cq += v * v;
        }
      }
      atomicAdd(&s_sum[col2], cs);
      atomicAdd(&s_sq[col2], cq);
    } else {
#pragma unroll
      for (int r = 0; r < 8; ++r) {
        if (row0 + lrb + r < N_PTS) {
          float v = c2[r] + bias2;
          int g = s_g[lrb + r];
          unsafeAtomicAdd(&sum2[g * D2 + col2], v);
          unsafeAtomicAdd(&sq2[g * D2 + col2], v * v);
        }
      }
    }
  }

  if (!FINAL) {
    __syncthreads();
    if (fast && tid < D2) {
      unsafeAtomicAdd(&sum2[gfirst * D2 + tid], s_sum[tid]);
      unsafeAtomicAdd(&sq2[gfirst * D2 + tid], s_sq[tid]);
    }
  }
}

// ---------------------------------------------------------------------------
// decode u32 keys -> float; empty segments (-inf) -> 0  (torch_scatter semantics)
// ---------------------------------------------------------------------------
__global__ __launch_bounds__(256) void k_fix(unsigned* __restrict__ outk,
                                             float* __restrict__ out) {
  int i = blockIdx.x * 256 + threadIdx.x;
  if (i < M_SUP * D2) {
    float v = funkey(outk[i]);
    out[i] = (v == -__builtin_inff()) ? 0.0f : v;
  }
}

// ---------------------------------------------------------------------------
extern "C" void kernel_launch(void* const* d_in, const int* in_sizes, int n_in,
                              void* d_out, int out_size, void* d_ws, size_t ws_size,
                              hipStream_t stream) {
  const float* x     = (const float*)d_in[0];
  const float* W1    = (const float*)d_in[1];
  const float* b1    = (const float*)d_in[2];
  const float* g1    = (const float*)d_in[3];
  const float* beta1 = (const float*)d_in[4];
  const float* W2    = (const float*)d_in[5];
  const float* b2    = (const float*)d_in[6];
  const float* g2    = (const float*)d_in[7];
  const float* beta2 = (const float*)d_in[8];
  const int*   gidx  = (const int*)d_in[9];
  const int*   sidx  = (const int*)d_in[10];

  float*    out  = (float*)d_out;
  unsigned* outk = (unsigned*)d_out;
  float*    ws   = (float*)d_ws;

  float* sum1 = ws;                       // [G,128]
  float* sq1  = ws + 1 * G_GR * D1;       // [G,128]
  float* sum2 = ws + 2 * G_GR * D1;       // [G,64]
  float* sq2  = sum2 + G_GR * D2;         // [G,64]
  float* A1c  = sq2 + G_GR * D2;          // [G,128]
  float* B1c  = A1c + G_GR * D1;          // [G,128]
  float* A2c  = B1c + G_GR * D1;          // [G,64]
  float* B2c  = A2c + G_GR * D2;          // [G,64]

  const int nOut  = M_SUP * D2;
  const int nbOut = (nOut + 255) / 256;
  const int nb    = (N_PTS + 127) / 128;

  k_init<<<nbOut, 256, 0, stream>>>(outk, ws);
  k_stats1<<<nb, 256, 0, stream>>>(x, W1, b1, gidx, sum1, sq1);
  k_finalize<<<(G_GR * D1 + 255) / 256, 256, 0, stream>>>(sum1, sq1, g1, beta1,
                                                          gidx, A1c, B1c, D1);
  k_fused<false><<<nb, 256, 0, stream>>>(x, W1, b1, W2, b2, gidx, sidx,
                                         A1c, B1c, A2c, B2c, sum2, sq2, outk);
  k_finalize<<<(G_GR * D2 + 255) / 256, 256, 0, stream>>>(sum2, sq2, g2, beta2,
                                                          gidx, A2c, B2c, D2);
  k_fused<true><<<nb, 256, 0, stream>>>(x, W1, b1, W2, b2, gidx, sidx,
                                        A1c, B1c, A2c, B2c, sum2, sq2, outk);
  k_fix<<<nbOut, 256, 0, stream>>>(outk, out);
}
